// OuterProductMean_16793322127901
// MI455X (gfx1250) — compile-verified, matrix operand order checked by reading, not run on hardware
//
#include <hip/hip_runtime.h>

typedef __attribute__((ext_vector_type(16))) __bf16 v16bf;
typedef __attribute__((ext_vector_type(8)))  float  v8f;

#define S_DIM  128
#define I_DIM  512
#define CM_DIM 256
#define C_DIM  32
#define D_DIM  32
#define CZ_DIM 128

#if defined(__AMDGCN__) && __has_builtin(__builtin_amdgcn_global_load_async_to_lds_b128) && \
    __has_builtin(__builtin_amdgcn_s_wait_asynccnt)
#define USE_ASYNC_LDS 1
// Parameter types per hipcc diagnostic: 4 x i32 vector pointers, global / shared AS.
typedef int v4i __attribute__((vector_size(4 * sizeof(int))));
typedef __attribute__((address_space(1))) v4i* async_gptr;
typedef __attribute__((address_space(3))) v4i* async_sptr;
#else
#define USE_ASYNC_LDS 0
#endif

__device__ __forceinline__ v8f wmma_bf16(v16bf a, v16bf b, v8f c) {
    // D = A x B + C, 16x16x32 bf16 -> f32
    return __builtin_amdgcn_wmma_f32_16x16x32_bf16(
        /*neg_a=*/false, a, /*neg_b=*/false, b,
        /*c_mod=*/(short)0, c, /*reuse_a=*/false, /*reuse_b=*/false);
}

__device__ __forceinline__ unsigned pack_bf16(float a, float b) {
    unsigned short ua = __builtin_bit_cast(unsigned short, (__bf16)a);
    unsigned short ub = __builtin_bit_cast(unsigned short, (__bf16)b);
    return (unsigned)ua | ((unsigned)ub << 16);
}

// ---------------- Kernel 0: convert weights to bf16 ----------------
// Wcomb[64][256]: rows 0..31 = Wa rows, 32..63 = Wb rows (K contiguous)
// Wo_bf[128][1024] = Wo * (1/S)   (folds the mean over S into stage-2 weights)
__global__ void opm_convert_weights(const float* __restrict__ Wa,
                                    const float* __restrict__ Wb,
                                    const float* __restrict__ Wo,
                                    __bf16* __restrict__ Wcomb,
                                    __bf16* __restrict__ Wo_bf) {
    int idx = blockIdx.x * 256 + threadIdx.x;
    if (idx < 64 * 256) {
        int n = idx >> 8, k = idx & 255;
        float v = (n < 32) ? Wa[n * 256 + k] : Wb[(n - 32) * 256 + k];
        Wcomb[idx] = (__bf16)v;
    }
    if (idx < 128 * 1024) {
        Wo_bf[idx] = (__bf16)(Wo[idx] * (1.0f / (float)S_DIM));
    }
}

// ---------------- Kernel 1: LayerNorm + A/B projections (WMMA) ----------------
// Each block: 128 rows of x (row g -> s = g/512, i = g%512).
// Output layouts (K = s contiguous, matching WMMA A / B^T fragment order):
//   aT[(i*32 + c) * 128 + s],  bT[(j*32 + d) * 128 + s]
__global__ void opm_ln_proj(const float* __restrict__ x,
                            const float* __restrict__ nw,
                            const float* __restrict__ nb,
                            const __bf16* __restrict__ Wcomb,
                            __bf16* __restrict__ aT,
                            __bf16* __restrict__ bT) {
    __shared__ __align__(16) __bf16 lnS[128 * 256];   // 64 KB

    const int tid  = threadIdx.x;
    const int w    = tid >> 5;
    const int lane = tid & 31;
    const int l15  = lane & 15;
    const int half = lane >> 4;
    const int r0   = blockIdx.x * 128;

    // per-lane norm params for elements [lane*8, lane*8+8)
    const float4 nw0 = *(const float4*)&nw[lane * 8];
    const float4 nw1 = *(const float4*)&nw[lane * 8 + 4];
    const float4 nb0 = *(const float4*)&nb[lane * 8];
    const float4 nb1 = *(const float4*)&nb[lane * 8 + 4];

    // ---- LayerNorm: each wave handles 16 rows, 32 lanes x 8 elems per row ----
    for (int rr = 0; rr < 16; ++rr) {
        const int m = w * 16 + rr;
        const float* row = x + (size_t)(r0 + m) * CM_DIM;
        const float4 v0 = *(const float4*)&row[lane * 8];
        const float4 v1 = *(const float4*)&row[lane * 8 + 4];

        float s  = v0.x + v0.y + v0.z + v0.w + v1.x + v1.y + v1.z + v1.w;
        float ss = v0.x*v0.x + v0.y*v0.y + v0.z*v0.z + v0.w*v0.w
                 + v1.x*v1.x + v1.y*v1.y + v1.z*v1.z + v1.w*v1.w;
        #pragma unroll
        for (int off = 16; off >= 1; off >>= 1) {
            s  += __shfl_xor(s,  off, 32);
            ss += __shfl_xor(ss, off, 32);
        }
        const float mu  = s * (1.0f / 256.0f);
        const float var = ss * (1.0f / 256.0f) - mu * mu;
        const float rs  = rsqrtf(var + 1e-5f);

        float o0 = (v0.x - mu) * rs * nw0.x + nb0.x;
        float o1 = (v0.y - mu) * rs * nw0.y + nb0.y;
        float o2 = (v0.z - mu) * rs * nw0.z + nb0.z;
        float o3 = (v0.w - mu) * rs * nw0.w + nb0.w;
        float o4 = (v1.x - mu) * rs * nw1.x + nb1.x;
        float o5 = (v1.y - mu) * rs * nw1.y + nb1.y;
        float o6 = (v1.z - mu) * rs * nw1.z + nb1.z;
        float o7 = (v1.w - mu) * rs * nw1.w + nb1.w;

        uint4 p;
        p.x = pack_bf16(o0, o1);
        p.y = pack_bf16(o2, o3);
        p.z = pack_bf16(o4, o5);
        p.w = pack_bf16(o6, o7);
        *(uint4*)&lnS[m * 256 + lane * 8] = p;
    }
    __syncthreads();

    // ---- Projection GEMM: M=128 (rows), N=64 (32 a-ch + 32 b-ch), K=256 ----
    // Wave w computes rows [w*16, w*16+16) against all 4 N-tiles.
    v8f acc[4] = {};
    #pragma unroll
    for (int k0 = 0; k0 < 256; k0 += 32) {
        const v16bf af = *(const v16bf*)&lnS[(w * 16 + l15) * 256 + k0 + half * 16];
        #pragma unroll
        for (int ct = 0; ct < 4; ++ct) {
            const v16bf bfr = *(const v16bf*)&Wcomb[(ct * 16 + l15) * 256 + k0 + half * 16];
            acc[ct] = wmma_bf16(af, bfr, acc[ct]);
        }
    }

    // ---- Scatter into K-major aT / bT ----
    #pragma unroll
    for (int ct = 0; ct < 4; ++ct) {
        const int n = ct * 16 + l15;
        #pragma unroll
        for (int e = 0; e < 8; ++e) {
            const int m = w * 16 + e + half * 8;
            const int g = r0 + m;
            const int s = g >> 9;        // / 512
            const int i = g & 511;       // % 512
            const __bf16 v = (__bf16)acc[ct][e];
            if (n < 32) aT[(size_t)(i * 32 + n)        * S_DIM + s] = v;
            else        bT[(size_t)(i * 32 + (n - 32)) * S_DIM + s] = v;
        }
    }
}

// ---------------- Kernel 2: fused outer-product-mean + Wo GEMM ----------------
// Block = one 4x4 (i,j) tile.
// Stage 1: O[(i,c),(j,d)] (128x128, K=S=128) via bf16 WMMA, LDS slabs.
// Stage 2: Z[cz, pair] = Wo_bf(128x1024) x Obuf^T(1024x16), K=1024.
__global__ void opm_main(const __bf16* __restrict__ aT,
                         const __bf16* __restrict__ bT,
                         const __bf16* __restrict__ Wo_bf,
                         const float* __restrict__ bo,
                         float* __restrict__ out) {
    __shared__ __align__(16) __bf16 smem[32768];     // 64 KB
    __bf16* As   = smem;             // 128x128 (32 KB)
    __bf16* Bs   = smem + 16384;     // 128x128 (32 KB)
    __bf16* Obuf = smem;             // 16x1024 (32 KB), overlays As after stage 1

    const int tid  = threadIdx.x;
    const int w    = tid >> 5;
    const int lane = tid & 31;
    const int l15  = lane & 15;
    const int half = lane >> 4;
    const int ti   = blockIdx.x >> 7;   // 0..127  -> i0 = ti*4
    const int tj   = blockIdx.x & 127;  // 0..127  -> j0 = tj*4

    // ---- Load contiguous 32 KB A and B slabs into LDS ----
    {
        const uint4* gA = (const uint4*)(aT + (size_t)ti * 16384);
        const uint4* gB = (const uint4*)(bT + (size_t)tj * 16384);
        uint4* sA = (uint4*)As;
        uint4* sB = (uint4*)Bs;
#if USE_ASYNC_LDS
        // CDNA5 async copy: global -> LDS without VGPR round trip (ASYNCcnt).
        #pragma unroll
        for (int t = 0; t < 8; ++t) {
            const int idx = t * 256 + tid;   // 2048 uint4 per slab
            __builtin_amdgcn_global_load_async_to_lds_b128(
                (async_gptr)(gA + idx), (async_sptr)(sA + idx),
                /*imm offset=*/0, /*cpol=*/0);
            __builtin_amdgcn_global_load_async_to_lds_b128(
                (async_gptr)(gB + idx), (async_sptr)(sB + idx),
                /*imm offset=*/0, /*cpol=*/0);
        }
        __builtin_amdgcn_s_wait_asynccnt(0);
#else
        #pragma unroll
        for (int t = 0; t < 8; ++t) {
            const int idx = t * 256 + tid;   // 2048 uint4 per slab
            sA[idx] = gA[idx];
            sB[idx] = gB[idx];
        }
#endif
    }
    __syncthreads();

    // ---- Stage 1: 128x128x128 GEMM; wave w owns M [ (w&3)*32, +32 ), N [ (w>>2)*64, +64 ) ----
    const int mbase = (w & 3) * 32;
    const int nbase = (w >> 2) * 64;
    v8f acc[2][4] = {};
    #pragma unroll
    for (int k0 = 0; k0 < S_DIM; k0 += 32) {
        v16bf af[2], bfr[4];
        #pragma unroll
        for (int r = 0; r < 2; ++r)
            af[r] = *(const v16bf*)&As[(mbase + r * 16 + l15) * S_DIM + k0 + half * 16];
        #pragma unroll
        for (int c = 0; c < 4; ++c)
            bfr[c] = *(const v16bf*)&Bs[(nbase + c * 16 + l15) * S_DIM + k0 + half * 16];
        #pragma unroll
        for (int r = 0; r < 2; ++r)
            #pragma unroll
            for (int c = 0; c < 4; ++c)
                acc[r][c] = wmma_bf16(af[r], bfr[c], acc[r][c]);
    }
    __syncthreads();   // all reads of As/Bs done before Obuf overlays As

    // ---- Repack O tile to Obuf[pair][cd] in bf16 ----
    #pragma unroll
    for (int r = 0; r < 2; ++r) {
        #pragma unroll
        for (int c = 0; c < 4; ++c) {
            const int nn = nbase + c * 16 + l15;
            const int pj = nn >> 5, d = nn & 31;
            #pragma unroll
            for (int e = 0; e < 8; ++e) {
                const int mm = mbase + r * 16 + e + half * 8;
                const int pi = mm >> 5, cc = mm & 31;
                Obuf[(pi * 4 + pj) * 1024 + cc * 32 + d] = (__bf16)acc[r][c][e];
            }
        }
    }
    __syncthreads();

    // ---- Stage 2: Z[128 x 16] = Wo_bf(128x1024) @ Obuf^T, wave w owns cz rows [w*16, +16) ----
    v8f acc2 = {};
    #pragma unroll 4
    for (int k0 = 0; k0 < 1024; k0 += 32) {
        const v16bf a2 = *(const v16bf*)&Wo_bf[(size_t)(w * 16 + l15) * 1024 + k0 + half * 16];
        const v16bf b2 = *(const v16bf*)&Obuf[l15 * 1024 + k0 + half * 16];
        acc2 = wmma_bf16(a2, b2, acc2);
    }

    // ---- Epilogue: add bias, write z[i, j, cz] fp32 ----
    {
        const int pair = l15;
        const int pi = pair >> 2, pj = pair & 3;
        const int i = ti * 4 + pi, j = tj * 4 + pj;
        float* op = out + ((size_t)i * I_DIM + j) * CZ_DIM;
        #pragma unroll
        for (int e = 0; e < 8; ++e) {
            const int cz = w * 16 + e + half * 8;
            op[cz] = acc2[e] + bo[cz];
        }
    }
}

extern "C" void kernel_launch(void* const* d_in, const int* in_sizes, int n_in,
                              void* d_out, int out_size, void* d_ws, size_t ws_size,
                              hipStream_t stream) {
    const float* x  = (const float*)d_in[0];
    const float* nw = (const float*)d_in[1];
    const float* nb = (const float*)d_in[2];
    const float* Wa = (const float*)d_in[3];
    const float* Wb = (const float*)d_in[4];
    const float* Wo = (const float*)d_in[5];
    const float* bo = (const float*)d_in[6];
    float* out = (float*)d_out;

    char* ws = (char*)d_ws;
    __bf16* aT    = (__bf16*)(ws);                                 // 16384*128 bf16 = 4 MB
    __bf16* bT    = (__bf16*)(ws + (size_t)(4u << 20));            // 4 MB
    __bf16* Wcomb = (__bf16*)(ws + (size_t)(8u << 20));            // 64*256 bf16 = 32 KB
    __bf16* Wo_bf = (__bf16*)(ws + (size_t)(8u << 20) + (64u << 10)); // 128*1024 bf16 = 256 KB

    opm_convert_weights<<<512, 256, 0, stream>>>(Wa, Wb, Wo, Wcomb, Wo_bf);
    opm_ln_proj<<<512, 256, 0, stream>>>(x, nw, nb, Wcomb, aT, bT);
    opm_main<<<16384, 256, 0, stream>>>(aT, bT, Wo_bf, bo, out);
}
